// EvoFill_76871324663930
// MI455X (gfx1250) — compile-verified, matrix operand order checked by reading, not run on hardware
//
#include <hip/hip_runtime.h>
#include <hip/hip_bf16.h>
#include <math.h>

// ---- model constants (from reference) ----
#define N_CATS   4
#define D_MODEL  256
#define D_STATE  64
#define HEADDIM  64
#define D_INNER  512
#define NHEADS   8
#define CONV_DIM 640            // D_INNER + 2*D_STATE
#define D_IN_PROJ 1160          // 2*D_INNER + 2*D_STATE + NHEADS
#define ZSTR     1216           // D_IN_PROJ padded to 19*64 for guard-free GEMM
#define D_CONV   4
#define BATCH    8
#define SEQLEN   8192
#define LC       2048           // chunk length
#define MROWS    (BATCH*LC)     // 16384 GEMM rows per chunk
#define EPSF     1e-5f

typedef __attribute__((ext_vector_type(16))) __bf16 v16bf;
typedef __attribute__((ext_vector_type(8)))  float  v8f;

__device__ __forceinline__ unsigned short f2bf(float f) {
  unsigned int u = __float_as_uint(f);
  u += 0x7FFFu + ((u >> 16) & 1u);          // round-to-nearest-even
  return (unsigned short)(u >> 16);
}
__device__ __forceinline__ float siluf(float x) { return x / (1.0f + __expf(-x)); }

// ============================================================
// bf16 WMMA GEMM: C[M,N] (+)= A[M,K] * B[K,N], A/B bf16 dense.
// tile 128x64x64, 8 waves (wave32), each wave 32x32 via 2x2 WMMA,
// two 32-k sub-steps per staged tile -> 8 v_wmma per barrier pair.
// LDS tiles are stored in WMMA-fragment-native order:
//  - within each 32-k block, k is bit-3<->bit-4 swapped so each
//    lane's 16-element fragment (32B) is 2 contiguous ds_load_b128.
//  - B is staged n-major (transposed at load time).
// ACCUM is a compile-time flag -> straight-line epilogue.
// Preconditions (all call sites): M%128==0, N%64==0, K%64==0.
// ============================================================
#define BM 128
#define BN 64
#define BK 64
#define LDT 72   // ushort stride per tile row (144B, 16B multiple)

__device__ __forceinline__ int kperm32(int k) {         // swap bits 3 and 4
  return (k & 7) | ((k & 8) << 1) | ((k & 16) >> 1);
}

template <int ACCUM>
__global__ __launch_bounds__(256) void gemm_bf16_wmma(
    const unsigned short* __restrict__ A,
    const unsigned short* __restrict__ Bg,
    float* __restrict__ C,
    int M, int N, int K)
{
  __shared__ __align__(16) unsigned short As[BM * LDT];
  __shared__ __align__(16) unsigned short Bs[BN * LDT];

  const int tid  = threadIdx.x;
  const int lane = tid & 31;
  const int wave = tid >> 5;
  const int wm   = wave >> 1;   // 0..3
  const int wn   = wave & 1;    // 0..1
  const int half = lane >> 4;
  const int l16  = lane & 15;
  const int blockM = blockIdx.y * BM;
  const int blockN = blockIdx.x * BN;

  v8f acc[2][2] = {};

  for (int kb = 0; kb < K; kb += BK) {
    // ---- A tile 128x64: 1024 chunks of 8 ushorts, 4 per thread ----
    #pragma unroll
    for (int it = 0; it < 4; ++it) {
      int chunk = tid + it * 256;
      int row = chunk >> 3;
      int col = (chunk & 7) << 3;                       // 0..56, 8-aligned
      int pcol = (col & 32) | kperm32(col & 31);
      const unsigned short* gp = A + (size_t)(blockM + row) * K + kb + col;
      uint4 val = *(const uint4*)gp;
      __builtin_prefetch(gp + BK, 0, 1);                // global_prefetch_b8
      *(uint4*)(&As[row * LDT + pcol]) = val;
    }
    // ---- B tile 64x64 -> transposed to Bs[n][perm(k)] ----
    #pragma unroll
    for (int it = 0; it < 2; ++it) {
      int chunk = tid + it * 256;
      int row = chunk >> 3;                             // k 0..63
      int col = (chunk & 7) << 3;                       // n base 0..56
      int pk  = (row & 32) | kperm32(row & 31);
      const unsigned short* gp = Bg + (size_t)(kb + row) * N + blockN + col;
      uint4 val = *(const uint4*)gp;
      __builtin_prefetch(gp + (size_t)BK * N, 0, 1);
      unsigned short tmp[8];
      *(uint4*)tmp = val;
      #pragma unroll
      for (int j = 0; j < 8; ++j)
        Bs[(col + j) * LDT + pk] = tmp[j];
    }
    __syncthreads();

    // ---- two 32-k sub-steps, 8 WMMA per staged tile ----
    #pragma unroll
    for (int sub = 0; sub < 2; ++sub) {
      union Frag { v16bf v; uint4 q[2]; } af[2], bfg[2];
      #pragma unroll
      for (int tm = 0; tm < 2; ++tm) {
        const unsigned short* p =
            &As[(wm * 32 + tm * 16 + l16) * LDT + sub * 32 + half * 16];
        af[tm].q[0] = *(const uint4*)(p);
        af[tm].q[1] = *(const uint4*)(p + 8);
      }
      #pragma unroll
      for (int tn = 0; tn < 2; ++tn) {
        const unsigned short* p =
            &Bs[(wn * 32 + tn * 16 + l16) * LDT + sub * 32 + half * 16];
        bfg[tn].q[0] = *(const uint4*)(p);
        bfg[tn].q[1] = *(const uint4*)(p + 8);
      }
      #pragma unroll
      for (int tm = 0; tm < 2; ++tm)
        #pragma unroll
        for (int tn = 0; tn < 2; ++tn)
          acc[tm][tn] = __builtin_amdgcn_wmma_f32_16x16x32_bf16(
              false, af[tm].v, false, bfg[tn].v, (short)0, acc[tm][tn],
              false, false);
    }
    __syncthreads();
  }

  // epilogue: D layout — VGPR v: M = v + half*8, N = lane&15 (no guards:
  // M,N padded to tile multiples at all call sites; ACCUM compile-time)
  #pragma unroll
  for (int tm = 0; tm < 2; ++tm)
    #pragma unroll
    for (int tn = 0; tn < 2; ++tn) {
      int r0 = blockM + wm * 32 + tm * 16 + (half << 3);
      int c  = blockN + wn * 32 + tn * 16 + l16;
      #pragma unroll
      for (int v = 0; v < 8; ++v) {
        size_t idx = (size_t)(r0 + v) * N + c;
        if (ACCUM) C[idx] += acc[tm][tn][v];
        else       C[idx]  = acc[tm][tn][v];
      }
    }
}

// ============================================================
// embedding + coord proj + LayerNorm : one block per token
// ============================================================
__global__ __launch_bounds__(256) void embed_ln_kernel(
    const int* __restrict__ x, const float* __restrict__ xc,
    const float* __restrict__ et, const float* __restrict__ cw,
    const float* __restrict__ cb, const float* __restrict__ g,
    const float* __restrict__ bt, float* __restrict__ h)
{
  __shared__ float red[256];
  int pos = blockIdx.x;              // b*SEQLEN + l
  int l   = pos & (SEQLEN - 1);
  int ch  = threadIdx.x;
  int idx = x[pos];
  if (idx < 0 || idx > N_CATS) idx = N_CATS;
  float e = et[idx * D_MODEL + ch] + cb[ch];
  #pragma unroll
  for (int j = 0; j < 4; ++j) e += xc[l * 4 + j] * cw[j * D_MODEL + ch];
  red[ch] = e; __syncthreads();
  for (int s = 128; s > 0; s >>= 1) { if (ch < s) red[ch] += red[ch + s]; __syncthreads(); }
  float mu = red[0] * (1.0f / 256.0f);
  __syncthreads();
  float d = e - mu;
  red[ch] = d * d; __syncthreads();
  for (int s = 128; s > 0; s >>= 1) { if (ch < s) red[ch] += red[ch + s]; __syncthreads(); }
  float var = red[0] * (1.0f / 256.0f);
  h[(size_t)pos * D_MODEL + ch] = d * rsqrtf(var + EPSF) * g[ch] + bt[ch];
}

// gather chunk rows (fp32) -> dense bf16 A matrix, K = D_MODEL
__global__ void gather_bf16_kernel(const float* __restrict__ src,
    unsigned short* __restrict__ dst, int srcStride, int srcOff)
{
  int t = blockIdx.x * blockDim.x + threadIdx.x;
  if (t >= MROWS * D_MODEL) return;
  int m = t >> 8, k = t & 255;
  int b = m >> 11, l = m & (LC - 1);
  dst[(size_t)m * D_MODEL + k] =
      f2bf(src[((size_t)(b * srcStride + srcOff + l)) * D_MODEL + k]);
}

__global__ void scatter_chunk_kernel(const float* __restrict__ src,
    float* __restrict__ dst, int chunkOff)
{
  int t = blockIdx.x * blockDim.x + threadIdx.x;
  if (t >= MROWS * D_MODEL) return;
  int m = t >> 8, k = t & 255;
  int b = m >> 11, l = m & (LC - 1);
  dst[((size_t)(b * SEQLEN + chunkOff + l)) * D_MODEL + k] = src[t];
}

__global__ void f32_to_bf16_kernel(const float* __restrict__ s,
                                   unsigned short* __restrict__ d, int n)
{
  int t = blockIdx.x * blockDim.x + threadIdx.x;
  if (t < n) d[t] = f2bf(s[t]);
}

// W_in (8 slices of 256 x 1160) -> bf16 padded to 256 x 1216 (zero pad)
__global__ void convert_pad_win_kernel(const float* __restrict__ s,
                                       unsigned short* __restrict__ d)
{
  int t = blockIdx.x * blockDim.x + threadIdx.x;
  if (t >= 8 * D_MODEL * ZSTR) return;
  int n  = t % ZSTR;
  int rk = t / ZSTR;                 // slice*256 + k
  d[t] = (n < D_IN_PROJ) ? f2bf(s[(size_t)rk * D_IN_PROJ + n])
                         : (unsigned short)0;
}

// depthwise conv (causal fwd / anti-causal rev) + SiLU
__global__ void conv_silu_kernel(const float* __restrict__ zx,
    const float* __restrict__ cw, const float* __restrict__ cb,
    float* __restrict__ xbc, int rev)
{
  int t = blockIdx.x * blockDim.x + threadIdx.x;
  if (t >= MROWS * CONV_DIM) return;
  int m = t / CONV_DIM, ch = t - m * CONV_DIM;
  int l = m & (LC - 1);
  float acc = cb[ch];
  #pragma unroll
  for (int k = 0; k < D_CONV; ++k) {
    int dl = rev ? (D_CONV - 1 - k) : (k - (D_CONV - 1));
    int ll = l + dl;
    if (ll >= 0 && ll < LC)
      acc += cw[ch * D_CONV + k] *
             zx[((size_t)(m - l + ll)) * ZSTR + D_INNER + ch];
  }
  xbc[(size_t)m * CONV_DIM + ch] = siluf(acc);
}

// dt = softplus(dt_raw + bias); dA = exp(-exp(A_log)*dt)
__global__ void dt_kernel(const float* __restrict__ zx,
    const float* __restrict__ dtb, const float* __restrict__ alog,
    float* __restrict__ dt, float* __restrict__ dA)
{
  int t = blockIdx.x * blockDim.x + threadIdx.x;
  if (t >= MROWS * NHEADS) return;
  int m = t >> 3, h = t & 7;
  float v = zx[(size_t)m * ZSTR + D_INNER + CONV_DIM + h] + dtb[h];
  float sp = (v > 20.f) ? v : log1pf(__expf(v));
  dt[t] = sp;
  dA[t] = __expf(-__expf(alog[h]) * sp);
}

// selective scan: block per (b,h); state (64x64) in registers, 16/lane
__global__ __launch_bounds__(256) void scan_kernel(
    const float* __restrict__ xbc, const float* __restrict__ dt,
    const float* __restrict__ dA, const float* __restrict__ Dp,
    float* __restrict__ y, int rev)
{
  int b = blockIdx.x >> 3;
  int h = blockIdx.x & 7;
  int tid = threadIdx.x;
  int p = tid >> 2, q = tid & 3;
  float st[16];
  #pragma unroll
  for (int j = 0; j < 16; ++j) st[j] = 0.f;
  float Dh = Dp[h];
  for (int t = 0; t < LC; ++t) {
    int l = rev ? (LC - 1 - t) : t;
    size_t m = (size_t)b * LC + l;
    float dtv = dt[m * NHEADS + h];
    float dav = dA[m * NHEADS + h];
    float xv  = xbc[m * CONV_DIM + h * HEADDIM + p];
    float cf  = dtv * xv;
    float partial = 0.f;
    #pragma unroll
    for (int j = 0; j < 16; ++j) {
      int n = q * 16 + j;
      float Bn = xbc[m * CONV_DIM + D_INNER + n];
      float Cn = xbc[m * CONV_DIM + D_INNER + D_STATE + n];
      float s  = dav * st[j] + cf * Bn;
      st[j] = s;
      partial += s * Cn;
    }
    partial += __shfl_xor(partial, 1, 32);
    partial += __shfl_xor(partial, 2, 32);
    if (q == 0) y[m * D_INNER + h * HEADDIM + p] = partial + Dh * xv;
  }
}

// y*silu(z), RMSNorm*nw -> bf16 for out_proj
__global__ __launch_bounds__(256) void gatednorm_kernel(
    const float* __restrict__ y, const float* __restrict__ zx,
    const float* __restrict__ nw, unsigned short* __restrict__ dst)
{
  __shared__ float red[256];
  size_t m = blockIdx.x;
  int t = threadIdx.x;
  float v0 = y[m * D_INNER + t]       * siluf(zx[m * ZSTR + t]);
  float v1 = y[m * D_INNER + t + 256] * siluf(zx[m * ZSTR + t + 256]);
  red[t] = v0 * v0 + v1 * v1; __syncthreads();
  for (int s = 128; s > 0; s >>= 1) { if (t < s) red[t] += red[t + s]; __syncthreads(); }
  float scale = rsqrtf(red[0] * (1.0f / 512.0f) + EPSF);
  dst[m * D_INNER + t]       = f2bf(v0 * scale * nw[t]);
  dst[m * D_INNER + t + 256] = f2bf(v1 * scale * nw[t + 256]);
}

// k=3 channel conv over L + exact GELU; block per (b,l), thread per out-ch
__global__ __launch_bounds__(256) void lpconv_gelu_kernel(
    const float* __restrict__ hs, const float* __restrict__ w,
    const float* __restrict__ bias, float* __restrict__ out)
{
  __shared__ float sh[3 * D_MODEL];
  int pos = blockIdx.x;              // b*SEQLEN + l
  int l   = pos & (SEQLEN - 1);
  int t   = threadIdx.x;
  #pragma unroll
  for (int j = 0; j < 3; ++j) {
    int ll = l + j - 1;
    sh[j * D_MODEL + t] =
        (ll >= 0 && ll < SEQLEN) ? hs[((size_t)(pos + j - 1)) * D_MODEL + t] : 0.f;
  }
  __syncthreads();
  float acc = bias[t];
  for (int i = 0; i < D_MODEL; ++i) {
    const float* wp = w + ((size_t)t * D_MODEL + i) * 3;  // lp_w[o][i][tap]
    acc += sh[i] * wp[0] + sh[D_MODEL + i] * wp[1] + sh[2 * D_MODEL + i] * wp[2];
  }
  out[(size_t)pos * D_MODEL + t] = 0.5f * acc * (1.0f + erff(acc * 0.70710678f));
}

__global__ void logits_kernel(const float* __restrict__ g,
    const float* __restrict__ ow, const float* __restrict__ ob,
    float* __restrict__ out)
{
  int pos = blockIdx.x * blockDim.x + threadIdx.x;
  if (pos >= BATCH * SEQLEN) return;
  const float* gp = g + (size_t)pos * D_MODEL;
  float a0 = 0.f, a1 = 0.f, a2 = 0.f, a3 = 0.f;
  for (int i = 0; i < D_MODEL; ++i) {
    float v = gp[i];
    a0 += v * ow[i];
    a1 += v * ow[D_MODEL + i];
    a2 += v * ow[2 * D_MODEL + i];
    a3 += v * ow[3 * D_MODEL + i];
  }
  out[pos * 4 + 0] = a0 + ob[0];
  out[pos * 4 + 1] = a1 + ob[1];
  out[pos * 4 + 2] = a2 + ob[2];
  out[pos * 4 + 3] = a3 + ob[3];
}

// ============================================================
extern "C" void kernel_launch(void* const* d_in, const int* in_sizes, int n_in,
                              void* d_out, int out_size, void* d_ws, size_t ws_size,
                              hipStream_t stream)
{
  (void)in_sizes; (void)n_in; (void)out_size; (void)ws_size;
  const int*   x     = (const int*)  d_in[0];
  const float* xc    = (const float*)d_in[1];
  const float* et    = (const float*)d_in[2];
  const float* cw    = (const float*)d_in[3];
  const float* cbv   = (const float*)d_in[4];
  const float* lng   = (const float*)d_in[5];
  const float* lnb   = (const float*)d_in[6];
  const float* Win   = (const float*)d_in[7];
  const float* Wout  = (const float*)d_in[8];
  const float* convw = (const float*)d_in[9];
  const float* convb = (const float*)d_in[10];
  const float* dtb   = (const float*)d_in[11];
  const float* alog  = (const float*)d_in[12];
  const float* Dpar  = (const float*)d_in[13];
  const float* nw    = (const float*)d_in[14];
  const float* lpw   = (const float*)d_in[15];
  const float* lpb   = (const float*)d_in[16];
  const float* ocw   = (const float*)d_in[17];
  const float* ocb   = (const float*)d_in[18];
  float* out = (float*)d_out;

  // ---- carve workspace (~335 MB) ----
  char* ws = (char*)d_ws;
  size_t off = 0;
  auto carve = [&](size_t bytes) -> char* {
    char* p = ws + off;
    off = (off + bytes + 255) & ~(size_t)255;
    return p;
  };
  float*          h_full  = (float*)carve((size_t)BATCH * SEQLEN * D_MODEL * 4);
  float*          hs_full = (float*)carve((size_t)BATCH * SEQLEN * D_MODEL * 4);
  unsigned short* Win_bf  = (unsigned short*)carve((size_t)8 * D_MODEL * ZSTR * 2);
  unsigned short* Wout_bf = (unsigned short*)carve((size_t)8 * D_INNER * D_MODEL * 2);
  unsigned short* A_bf    = (unsigned short*)carve((size_t)MROWS * D_INNER * 2);
  float*          zx      = (float*)carve((size_t)MROWS * ZSTR * 4);
  float*          xbc     = (float*)carve((size_t)MROWS * CONV_DIM * 4);
  float*          dtbuf   = (float*)carve((size_t)MROWS * NHEADS * 4);
  float*          dabuf   = (float*)carve((size_t)MROWS * NHEADS * 4);
  float*          ybuf    = (float*)carve((size_t)MROWS * D_INNER * 4);
  float*          hcnew   = (float*)carve((size_t)MROWS * D_MODEL * 4);
  float*          gbuf    = zx;   // reused after chunk loop (64MB <= 79.7MB)

  // ---- stage 0: weights -> bf16 (W_in zero-padded to ZSTR), embed+LN ----
  {
    int n1 = 8 * D_MODEL * ZSTR;
    convert_pad_win_kernel<<<(n1 + 255) / 256, 256, 0, stream>>>(Win, Win_bf);
    int n2 = 8 * D_INNER * D_MODEL;
    f32_to_bf16_kernel<<<(n2 + 255) / 256, 256, 0, stream>>>(Wout, Wout_bf, n2);
    embed_ln_kernel<<<BATCH * SEQLEN, 256, 0, stream>>>(x, xc, et, cw, cbv, lng, lnb, h_full);
  }

  const int starts[4] = {0, 1844, 3688, 5532};

  for (int c = 0; c < 4; ++c) {
    for (int ly = 0; ly < 2; ++ly) {
      int sl = c * 2 + ly;
      // input rows -> bf16 (layer0 from h_full chunk, layer1 from hcnew)
      if (ly == 0)
        gather_bf16_kernel<<<MROWS, 256, 0, stream>>>(h_full, A_bf, SEQLEN, starts[c]);
      else
        gather_bf16_kernel<<<MROWS, 256, 0, stream>>>(hcnew, A_bf, LC, 0);

      // in_proj GEMM: 16384 x 1216 x 256 (N padded, guard-free)
      gemm_bf16_wmma<0><<<dim3(ZSTR / BN, MROWS / BM), 256, 0, stream>>>(
          A_bf, Win_bf + (size_t)sl * D_MODEL * ZSTR, zx,
          MROWS, ZSTR, D_MODEL);

      for (int dir = 0; dir < 2; ++dir) {
        int sd = sl * 2 + dir;
        conv_silu_kernel<<<(MROWS * CONV_DIM + 255) / 256, 256, 0, stream>>>(
            zx, convw + (size_t)sd * CONV_DIM * D_CONV,
            convb + (size_t)sd * CONV_DIM, xbc, dir);
        dt_kernel<<<(MROWS * NHEADS + 255) / 256, 256, 0, stream>>>(
            zx, dtb + (size_t)sd * NHEADS, alog + (size_t)sd * NHEADS, dtbuf, dabuf);
        scan_kernel<<<BATCH * NHEADS, 256, 0, stream>>>(
            xbc, dtbuf, dabuf, Dpar + (size_t)sd * NHEADS, ybuf, dir);
        gatednorm_kernel<<<MROWS, 256, 0, stream>>>(
            ybuf, zx, nw + (size_t)sd * D_INNER, A_bf);
        // out_proj GEMM: 16384 x 256 x 512, accumulate across directions
        if (dir == 0)
          gemm_bf16_wmma<0><<<dim3(D_MODEL / BN, MROWS / BM), 256, 0, stream>>>(
              A_bf, Wout_bf + (size_t)sl * D_INNER * D_MODEL, hcnew,
              MROWS, D_MODEL, D_INNER);
        else
          gemm_bf16_wmma<1><<<dim3(D_MODEL / BN, MROWS / BM), 256, 0, stream>>>(
              A_bf, Wout_bf + (size_t)sl * D_INNER * D_MODEL, hcnew,
              MROWS, D_MODEL, D_INNER);
      }
    }
    scatter_chunk_kernel<<<MROWS, 256, 0, stream>>>(hcnew, hs_full, c * LC);
  }

  // ---- head: k=3 channel conv + GELU, then logits ----
  lpconv_gelu_kernel<<<BATCH * SEQLEN, 256, 0, stream>>>(hs_full, lpw, lpb, gbuf);
  logits_kernel<<<(BATCH * SEQLEN + 255) / 256, 256, 0, stream>>>(gbuf, ocw, ocb, out);
}
